// HistoLoss_79645873537299
// MI455X (gfx1250) — compile-verified, matrix operand order checked by reading, not run on hardware
//
#include <hip/hip_runtime.h>

typedef __attribute__((ext_vector_type(16))) _Float16 v16h;
typedef __attribute__((ext_vector_type(8)))  float    v8f;

#define N_ROWS   131072          // B*T = 128*1024
#define F_DIM    16
#define NBINS    64
#define SLAB     256             // rows per staged slab
#define N_WG     256             // workgroups for the histogram kernel
#define SPW      (N_ROWS / SLAB / N_WG)   // = 2 slabs per workgroup
#define N_COLS   (F_DIM * NBINS) // 1024 (f,b) columns

#define USE_ASYNC 1              // CDNA5 async global->LDS staging (inline asm)

__device__ __forceinline__ _Float16 indf(float v, float loc, float halfw) {
    // reference: (relu(delta/2 - |x-loc|) > 0) == |x-loc| < delta/2 (strict)
    return (_Float16)((__builtin_fabsf(v - loc) < halfw) ? 1.0f : 0.0f);
}

__global__ void zero_counts(float* counts) {
    int i = blockIdx.x * blockDim.x + threadIdx.x;
    if (i < N_COLS) counts[i] = 0.0f;
}

// Stage one 256-row slab into LDS in TRANSPOSED layout T[f][row].
// Async path: 16x global_load_async_to_lds_b32 per thread; the async unit
// performs the transpose via per-lane LDS target addresses (no DS issue cost,
// tracked by ASYNCcnt, overlaps with WMMA compute on the other buffer).
__device__ __forceinline__ void stage_slab(float* Tb, const float* x,
                                           long rowBase, int t) {
#if USE_ASYNC
    #pragma unroll
    for (int f = 0; f < F_DIM; ++f) {
        unsigned int       ldsOff = (unsigned int)(size_t)(&Tb[f * SLAB + t]);
        unsigned long long gaddr  = (unsigned long long)(size_t)(x + (rowBase + t) * F_DIM + f);
        asm volatile("global_load_async_to_lds_b32 %0, %1, off"
                     :: "v"(ldsOff), "v"(gaddr) : "memory");
    }
#else
    const float4* src = (const float4*)(x + (rowBase + t) * F_DIM);
    float4 r0 = src[0], r1 = src[1], r2 = src[2], r3 = src[3];
    Tb[ 0*SLAB + t] = r0.x; Tb[ 1*SLAB + t] = r0.y;
    Tb[ 2*SLAB + t] = r0.z; Tb[ 3*SLAB + t] = r0.w;
    Tb[ 4*SLAB + t] = r1.x; Tb[ 5*SLAB + t] = r1.y;
    Tb[ 6*SLAB + t] = r1.z; Tb[ 7*SLAB + t] = r1.w;
    Tb[ 8*SLAB + t] = r2.x; Tb[ 9*SLAB + t] = r2.y;
    Tb[10*SLAB + t] = r2.z; Tb[11*SLAB + t] = r2.w;
    Tb[12*SLAB + t] = r3.x; Tb[13*SLAB + t] = r3.y;
    Tb[14*SLAB + t] = r3.z; Tb[15*SLAB + t] = r3.w;
#endif
}

__device__ __forceinline__ void wait_stage() {
#if USE_ASYNC
    asm volatile("s_wait_asynccnt 0x0" ::: "memory");
#endif
}

__global__ __launch_bounds__(256) void hist_wmma(
    const float* __restrict__ x,       // (N_ROWS, F_DIM)
    const float* __restrict__ locs,    // (F_DIM, NBINS)
    const float* __restrict__ deltas,  // (F_DIM,)
    float* __restrict__ counts)        // (F_DIM*NBINS,) fp32 counts
{
    __shared__ __align__(16) float T0[F_DIM * SLAB];   // 16 KB, [f][row]
    __shared__ __align__(16) float T1[F_DIM * SLAB];   // 16 KB, double buffer

    const int t      = threadIdx.x;
    const int lane   = t & 31;
    const int wave   = t >> 5;
    const int colSel = lane & 15;        // column within 16-bin tile
    const int kGrp   = (lane >> 4) << 4; // lanes 0-15 -> K 0..15, lanes 16-31 -> K 16..31

    v16h ones;
    #pragma unroll
    for (int h = 0; h < 16; ++h) ones[h] = (_Float16)1.0f;

    // per-wave tile parameters (8 tiles/wave; tile -> f = tile/4, bins (tile%4)*16..+15)
    float locv[8], halfv[8];
    int   colv[8];
    #pragma unroll
    for (int i = 0; i < 8; ++i) {
        const int tile = wave * 8 + i;
        const int f    = tile >> 2;
        const int b    = ((tile & 3) << 4) + colSel;
        colv[i]  = f * NBINS + b;
        locv[i]  = locs[colv[i]];
        halfv[i] = 0.5f * deltas[f];
    }

    v8f acc[8];
    {
        v8f z = {};
        #pragma unroll
        for (int i = 0; i < 8; ++i) acc[i] = z;
    }

    // prologue: kick async stage of slab 0
    stage_slab(T0, x, (long)blockIdx.x * SLAB, t);

    #pragma unroll
    for (int s = 0; s < SPW; ++s) {
        float* cur = (s & 1) ? T1 : T0;
        wait_stage();        // my async loads into cur are complete
        __syncthreads();     // everyone's are (and all reads of the other buffer done)

        if (s + 1 < SPW) {   // overlap next slab's async staging with compute
            float* nxt = (s & 1) ? T0 : T1;
            stage_slab(nxt, x, (long)(blockIdx.x + (s + 1) * N_WG) * SLAB, t);
        }

        #pragma unroll
        for (int i = 0; i < 8; ++i) {
            const int    f   = (wave * 8 + i) >> 2;
            const float  loc = locv[i];
            const float  hw  = halfv[i];
            const float4* base = (const float4*)&cur[f * SLAB];
            #pragma unroll
            for (int chunk = 0; chunk < 8; ++chunk) {
                // 16 rows for this lane's K-group: 4x ds_load_b128 (broadcast over 16 lanes)
                const float4* P = base + ((chunk * 32 + kGrp) >> 2);
                float4 q0 = P[0], q1 = P[1], q2 = P[2], q3 = P[3];
                v16h bv;
                bv[ 0] = indf(q0.x, loc, hw); bv[ 1] = indf(q0.y, loc, hw);
                bv[ 2] = indf(q0.z, loc, hw); bv[ 3] = indf(q0.w, loc, hw);
                bv[ 4] = indf(q1.x, loc, hw); bv[ 5] = indf(q1.y, loc, hw);
                bv[ 6] = indf(q1.z, loc, hw); bv[ 7] = indf(q1.w, loc, hw);
                bv[ 8] = indf(q2.x, loc, hw); bv[ 9] = indf(q2.y, loc, hw);
                bv[10] = indf(q2.z, loc, hw); bv[11] = indf(q2.w, loc, hw);
                bv[12] = indf(q3.x, loc, hw); bv[13] = indf(q3.y, loc, hw);
                bv[14] = indf(q3.z, loc, hw); bv[15] = indf(q3.w, loc, hw);
                // D = ones(16x32) * bv(32x16) + C : every row of D = column sums
                acc[i] = __builtin_amdgcn_wmma_f32_16x16x32_f16(
                             false, ones, false, bv, (short)0, acc[i], false, false);
            }
        }
        // no trailing barrier needed: next iteration's wait+barrier orders
        // buffer reuse against all waves' compute
    }

    // flush: M=0 row of D is acc[i][0] in lanes 0..15 (N = lane); exact integer f32
    #pragma unroll
    for (int i = 0; i < 8; ++i) {
        if (lane < 16) atomicAdd(&counts[colv[i]], acc[i][0]);
    }
}

__global__ __launch_bounds__(256) void histo_loss(
    const float* __restrict__ counts,
    const float* __restrict__ deltas,
    const float* __restrict__ dens,
    float* __restrict__ out)
{
    __shared__ float red[256];
    const int t = threadIdx.x;
    float s = 0.0f;
    for (int i = t; i < N_COLS; i += 256) {
        const int   f       = i >> 6;
        const float density = counts[i] * (1.0f / (float)N_ROWS) / deltas[f];
        s += __builtin_fabsf(density - dens[i]);
    }
    red[t] = s;
    __syncthreads();
    for (int off = 128; off > 0; off >>= 1) {
        if (t < off) red[t] += red[t + off];
        __syncthreads();
    }
    if (t == 0) out[0] = red[0] * (1.0f / (float)N_COLS);  // mean_b then mean_f
}

extern "C" void kernel_launch(void* const* d_in, const int* in_sizes, int n_in,
                              void* d_out, int out_size, void* d_ws, size_t ws_size,
                              hipStream_t stream) {
    const float* x      = (const float*)d_in[0];  // x_fake (131072*16)
    const float* locs   = (const float*)d_in[1];  // (16*64)
    const float* deltas = (const float*)d_in[2];  // (16)
    const float* dens   = (const float*)d_in[3];  // (16*64)
    float* counts = (float*)d_ws;                 // 1024 floats scratch
    float* out    = (float*)d_out;

    zero_counts<<<4, 256, 0, stream>>>(counts);
    hist_wmma<<<N_WG, 256, 0, stream>>>(x, locs, deltas, counts);
    histo_loss<<<1, 256, 0, stream>>>(counts, deltas, dens, out);
}
